// MCAttention_25984552140883
// MI455X (gfx1250) — matrix operand order repaired and checked by reading, not compiled, not run on hardware
//
#include <hip/hip_runtime.h>
#include <math.h>

// ---------------------------------------------------------------------------
// Types for CDNA5 WMMA (wave32): V_WMMA_F32_16X16X32_F16
// ---------------------------------------------------------------------------
typedef __attribute__((ext_vector_type(16))) _Float16 v16h;
typedef __attribute__((ext_vector_type(8)))  _Float16 v8h;
typedef __attribute__((ext_vector_type(8)))  float    v8f;

union AFrag { v16h v; v8h h[2]; };

__device__ __forceinline__ v8f v8f_zero() {
  v8f z = {0.f,0.f,0.f,0.f,0.f,0.f,0.f,0.f};
  return z;
}

__device__ __forceinline__ v8f wmma_f16(v16h a, v16h b, v8f c) {
  // D = A(16x32 f16) * B(32x16 f16) + C(16x16 f32)
  return __builtin_amdgcn_wmma_f32_16x16x32_f16(false, a, false, b, (short)0, c, false, false);
}

static constexpr int B_ = 16;   // batch
static constexpr int H_ = 8;    // heads

// ---------------------------------------------------------------------------
// f32 -> f16 convert (weights)
// ---------------------------------------------------------------------------
__global__ void k_convert(const float* __restrict__ src, _Float16* __restrict__ dst, int n) {
  int i = blockIdx.x * blockDim.x + threadIdx.x;
  if (i < n) dst[i] = (_Float16)src[i];
}

// ---------------------------------------------------------------------------
// flatten: b c (h s1) (w s2) -> (b*h*w, s1*s2*c), f32 -> f16
// ---------------------------------------------------------------------------
__global__ void k_flatten(const float* __restrict__ x, _Float16* __restrict__ X,
                          int C, int hh, int ww, int s1, int s2, int Ntok, int E) {
  long i = (long)blockIdx.x * blockDim.x + threadIdx.x;
  long total = (long)B_ * Ntok * E;
  if (i >= total) return;
  int e = (int)(i % E); long t = i / E;
  int bb = (int)(t / Ntok); int p = (int)(t % Ntok);
  int hy = p / ww, wx = p % ww;
  int i1 = e / (s2 * C); int rem = e % (s2 * C); int i2 = rem / C; int c = rem % C;
  int Hf = hh * s1, Wf = ww * s2;
  long idx = (((long)bb * C + c) * Hf + hy * s1 + i1) * Wf + wx * s2 + i2;
  X[i] = (_Float16)x[idx];
}

// ---------------------------------------------------------------------------
// GEMM core: one wave computes a 32x64 tile of A(MxK f16) @ W^T,
// W stored (N,K) row-major.  2 A-frags x 4 B-frags -> 8 WMMAs per 32-K chunk.
// Explicit one-chunk-ahead double buffering (K/32 is even for all scales:
// 6/12/24).  All loads use uniform SGPR base + 32-bit VGPR offset to
// minimize address VGPR pressure.
// acc[j]   : rows m0..m0+15,  cols n0+16j..+15
// acc[4+j] : rows m0+16..+31, cols n0+16j..+15
// ---------------------------------------------------------------------------
struct Chunk { AFrag a0, a1; v16h b[4]; };

__device__ __forceinline__ void load_chunk(Chunk& ch,
                                           const _Float16* __restrict__ A,
                                           const _Float16* __restrict__ W,
                                           unsigned a0, unsigned a1,
                                           unsigned b0, unsigned b1,
                                           unsigned b2, unsigned b3,
                                           unsigned k0) {
  ch.a0.h[0] = *(const v8h*)(A + a0 + k0);
  ch.a0.h[1] = *(const v8h*)(A + a0 + k0 + 8);
  ch.a1.h[0] = *(const v8h*)(A + a1 + k0);
  ch.a1.h[1] = *(const v8h*)(A + a1 + k0 + 8);
  ch.b[0] = *(const v16h*)(W + b0 + k0);
  ch.b[1] = *(const v16h*)(W + b1 + k0);
  ch.b[2] = *(const v16h*)(W + b2 + k0);
  ch.b[3] = *(const v16h*)(W + b3 + k0);
}

__device__ __forceinline__ void mm_chunk(const Chunk& ch, v8f acc[8]) {
#pragma unroll
  for (int j = 0; j < 4; j++) {
    acc[j]     = wmma_f16(ch.a0.v, ch.b[j], acc[j]);
    acc[4 + j] = wmma_f16(ch.a1.v, ch.b[j], acc[4 + j]);
  }
}

__device__ __forceinline__ void gemm_core(const _Float16* __restrict__ A,
                                          const _Float16* __restrict__ W,
                                          int K, int m0, int n0,
                                          int hi, int lo, v8f acc[8]) {
#pragma unroll
  for (int j = 0; j < 8; j++) acc[j] = v8f_zero();
  unsigned uK = (unsigned)K;
  unsigned a0 = (unsigned)(m0 + lo) * uK + 16u * hi;
  unsigned a1 = a0 + 16u * uK;
  unsigned b0 = (unsigned)(n0 + lo) * uK + 16u * hi;
  unsigned b1 = b0 + 16u * uK;
  unsigned b2 = b0 + 32u * uK;
  unsigned b3 = b0 + 48u * uK;

  Chunk c0, c1;
  load_chunk(c0, A, W, a0, a1, b0, b1, b2, b3, 0);
  int nch = K >> 5;          // even (6, 12, or 24)
  unsigned k = 32;
  for (int it = 0; it < (nch - 2) >> 1; it++) {
    load_chunk(c1, A, W, a0, a1, b0, b1, b2, b3, k);
    mm_chunk(c0, acc);
    load_chunk(c0, A, W, a0, a1, b0, b1, b2, b3, k + 32);
    mm_chunk(c1, acc);
    k += 64;
  }
  load_chunk(c1, A, W, a0, a1, b0, b1, b2, b3, k);
  mm_chunk(c0, acc);
  mm_chunk(c1, acc);
}

// ---------------------------------------------------------------------------
// GEMM: out(MxN f16) = A @ W^T + bias.  M ceil-tiled to 32 (ghost-row loads
// stay inside padded workspace; stores guarded).
// ---------------------------------------------------------------------------
__global__ void k_gemm_f16(const _Float16* __restrict__ A, const _Float16* __restrict__ W,
                           const float* __restrict__ bias, _Float16* __restrict__ out,
                           int M, int K, int N) {
  int wid  = (blockIdx.x * blockDim.x + threadIdx.x) >> 5;
  int lane = threadIdx.x & 31;
  int tilesN = N >> 6;
  int tilesM = (M + 31) >> 5;
  if (wid >= tilesM * tilesN) return;
  int mt = wid / tilesN, nt = wid % tilesN;
  int m0 = mt * 32, n0 = nt * 64;
  int hi = lane >> 4, lo = lane & 15;

  v8f acc[8];
  gemm_core(A, W, K, m0, n0, hi, lo, acc);

#pragma unroll
  for (int g = 0; g < 2; g++) {
#pragma unroll
    for (int j = 0; j < 4; j++) {
      int col = n0 + 16 * j + lo;
      float bv = bias[col];
#pragma unroll
      for (int r = 0; r < 8; r++) {
        int row = m0 + 16 * g + r + 8 * hi;
        if (row < M)
          out[(long)row * N + col] = (_Float16)(acc[4 * g + j][r] + bv);
      }
    }
  }
}

// ---------------------------------------------------------------------------
// Output-projection GEMM with fused unflatten + residual, f32 store to d_out.
// ---------------------------------------------------------------------------
__global__ void k_gemm_out(const _Float16* __restrict__ A, const _Float16* __restrict__ W,
                           const float* __restrict__ bias, const float* __restrict__ detail,
                           float* __restrict__ out,
                           int M, int K, int N,
                           int C, int hh, int ww, int s1, int s2, int Ntok) {
  int wid  = (blockIdx.x * blockDim.x + threadIdx.x) >> 5;
  int lane = threadIdx.x & 31;
  int tilesN = N >> 6;
  int tilesM = (M + 31) >> 5;
  if (wid >= tilesM * tilesN) return;
  int mt = wid / tilesN, nt = wid % tilesN;
  int m0 = mt * 32, n0 = nt * 64;
  int hi = lane >> 4, lo = lane & 15;

  v8f acc[8];
  gemm_core(A, W, K, m0, n0, hi, lo, acc);

  int Hf = hh * s1, Wf = ww * s2;
#pragma unroll
  for (int g = 0; g < 2; g++) {
#pragma unroll
    for (int j = 0; j < 4; j++) {
      int col = n0 + 16 * j + lo;
      float bv = bias[col];
      int i1 = col / (s2 * C); int rem = col % (s2 * C); int i2 = rem / C; int c = rem % C;
#pragma unroll
      for (int r = 0; r < 8; r++) {
        int row = m0 + 16 * g + r + 8 * hi;      // global token
        if (row < M) {
          int bb = row / Ntok; int p = row % Ntok;
          int hy = p / ww, wx = p % ww;
          long idx = (((long)bb * C + c) * Hf + hy * s1 + i1) * Wf + wx * s2 + i2;
          out[idx] = detail[idx] + acc[4 * g + j][r] + bv;
        }
      }
    }
  }
}

// ---------------------------------------------------------------------------
// Scatter proj (B*N, E) into padded per-head layouts (gather form, pads = 0).
// Q/K: dst[b][h][np][jp],  V: dst[b][h][jp][np] (transposed for PV GEMM).
// ---------------------------------------------------------------------------
__global__ void k_scatter_rows(const _Float16* __restrict__ proj, _Float16* __restrict__ dst,
                               int Np, int dp, int N, int d, int E) {
  long i = (long)blockIdx.x * blockDim.x + threadIdx.x;
  long total = (long)B_ * H_ * Np * dp;
  if (i >= total) return;
  int jp = (int)(i % dp); long t = i / dp;
  int np = (int)(t % Np); t /= Np;
  int hh = (int)(t % H_); int bb = (int)(t / H_);
  _Float16 v = (_Float16)0.f;
  if (np < N && jp < d) v = proj[((long)bb * N + np) * E + hh * d + jp];
  dst[i] = v;
}

__global__ void k_scatter_vt(const _Float16* __restrict__ proj, _Float16* __restrict__ dst,
                             int Np, int dp, int N, int d, int E) {
  long i = (long)blockIdx.x * blockDim.x + threadIdx.x;
  long total = (long)B_ * H_ * dp * Np;
  if (i >= total) return;
  int np = (int)(i % Np); long t = i / Np;
  int jp = (int)(t % dp); t /= dp;
  int hh = (int)(t % H_); int bb = (int)(t / H_);
  _Float16 v = (_Float16)0.f;
  if (np < N && jp < d) v = proj[((long)bb * N + np) * E + hh * d + jp];
  dst[i] = v;
}

// ---------------------------------------------------------------------------
// Flash attention, one wave = 16 queries, key blocks of 32, templated on
// padded head dim DP (32/64/96).  8 waves/block, all uniform -> __syncthreads
// is safe.  Q:(bh,Npq,DP)  K:(bh,Npk,DP)  VT:(bh,DP,Npk)  out:(B*N,E) f16.
// All hot loads use uniform base + 32-bit offset; 16*DP strides are
// compile-time so they fold into load immediate offsets.
// ---------------------------------------------------------------------------
template <int DP>
__global__ void k_attn(const _Float16* __restrict__ Q, const _Float16* __restrict__ Km,
                       const _Float16* __restrict__ VT, _Float16* __restrict__ out,
                       int N, int Npq, int Npk, int d, int E, float scale) {
  constexpr int CH = DP / 32;  // K-chunks for S = Q K^T
  constexpr int DT = DP / 16;  // N-tiles for O = P V
  __shared__ _Float16 Plds[8][16 * 32];

  int wv   = threadIdx.x >> 5;
  int lane = threadIdx.x & 31;
  int hi = lane >> 4, lo = lane & 15;
  int bh = blockIdx.z;
  int bb = bh / H_, hh = bh % H_;
  int q0 = (blockIdx.x * 8 + wv) * 16;

  _Float16* P = &Plds[wv][0];

  // 32-bit offsets; uniform parts are scalarized by the compiler
  unsigned qoff  = (unsigned)bh * Npq * DP + (unsigned)(q0 + lo) * DP + 16u * hi;
  unsigned kbase = (unsigned)bh * Npk * DP + (unsigned)lo * DP + 16u * hi;
  unsigned vbase = (unsigned)bh * DP * Npk + (unsigned)lo * Npk + 16u * hi;
  unsigned vrow[DT];
#pragma unroll
  for (int j = 0; j < DT; j++) vrow[j] = vbase + (unsigned)(16 * j) * Npk;

  float m[8], l[8];
#pragma unroll
  for (int r = 0; r < 8; r++) { m[r] = -1e30f; l[r] = 0.f; }
  v8f accO[DT];
#pragma unroll
  for (int j = 0; j < DT; j++) accO[j] = v8f_zero();

  int nkb = Npk / 32;

  for (int kb = 0; kb < nkb; kb++) {
    unsigned kk = kbase + (unsigned)kb * 32u * DP;   // kb*32*DP uniform -> SGPR
    v8f s0 = v8f_zero(), s1 = v8f_zero();
#pragma unroll
    for (int c = 0; c < CH; c++) {
      AFrag a;
      a.h[0] = *(const v8h*)(Q + qoff + c * 32);
      a.h[1] = *(const v8h*)(Q + qoff + c * 32 + 8);
      v16h b0 = *(const v16h*)(Km + kk + c * 32);
      v16h b1 = *(const v16h*)(Km + kk + 16u * DP + c * 32);  // 16*DP: imm-foldable
      s0 = wmma_f16(a.v, b0, s0);
      s1 = wmma_f16(a.v, b1, s1);
    }
    bool v0 = (kb * 32 + lo) < N;
    bool v1 = (kb * 32 + 16 + lo) < N;
    float corr[8];
#pragma unroll
    for (int r = 0; r < 8; r++) {
      float x0 = v0 ? s0[r] * scale : -1e30f;
      float x1 = v1 ? s1[r] * scale : -1e30f;
      float t = fmaxf(x0, x1);
      t = fmaxf(t, __shfl_xor(t, 1, 16));
      t = fmaxf(t, __shfl_xor(t, 2, 16));
      t = fmaxf(t, __shfl_xor(t, 4, 16));
      t = fmaxf(t, __shfl_xor(t, 8, 16));
      float mn = fmaxf(m[r], t);
      corr[r] = __expf(m[r] - mn);
      m[r] = mn;
      float p0 = __expf(x0 - mn);
      float p1 = __expf(x1 - mn);
      float ps = p0 + p1;
      ps += __shfl_xor(ps, 1, 16);
      ps += __shfl_xor(ps, 2, 16);
      ps += __shfl_xor(ps, 4, 16);
      ps += __shfl_xor(ps, 8, 16);
      l[r] = l[r] * corr[r] + ps;
      P[(r + 8 * hi) * 32 + lo]      = (_Float16)p0;
      P[(r + 8 * hi) * 32 + 16 + lo] = (_Float16)p1;
    }
#pragma unroll
    for (int j = 0; j < DT; j++)
#pragma unroll
      for (int r = 0; r < 8; r++) accO[j][r] *= corr[r];

    __syncthreads();                 // C-layout P -> LDS visible
    AFrag pa;
    pa.h[0] = *(const v8h*)(P + lo * 32 + 16 * hi);
    pa.h[1] = *(const v8h*)(P + lo * 32 + 8 + 16 * hi);
#pragma unroll
    for (int j = 0; j < DT; j++) {
      v16h b = *(const v16h*)(VT + vrow[j] + (unsigned)kb * 32u);
      accO[j] = wmma_f16(pa.v, b, accO[j]);
    }
    __syncthreads();                 // protect P before next overwrite
  }

  float inv[8];
#pragma unroll
  for (int r = 0; r < 8; r++) inv[r] = (l[r] > 0.f) ? 1.f / l[r] : 0.f;
#pragma unroll
  for (int j = 0; j < DT; j++) {
    int dim = j * 16 + lo;
    if (dim < d) {
#pragma unroll
      for (int r = 0; r < 8; r++) {
        int qr = q0 + r + 8 * hi;
        if (qr < N)
          out[((long)bb * N + qr) * E + hh * d + dim] = (_Float16)(accO[j][r] * inv[r]);
      }
    }
  }
}

// ---------------------------------------------------------------------------
// Host launcher
// ---------------------------------------------------------------------------
struct Cfg { int C, hh, ww, s1, s2, N, E, d, dp, Npq, Npk; };

extern "C" void kernel_launch(void* const* d_in, const int* in_sizes, int n_in,
                              void* d_out, int out_size, void* d_ws, size_t ws_size,
                              hipStream_t stream) {
  (void)in_sizes; (void)n_in; (void)out_size; (void)ws_size;

  static const Cfg cfg[4] = {
    { 96, 28, 28, 2, 2, 784, 384, 48, 64, 896, 800},
    {192, 28, 28, 1, 1, 784, 192, 24, 32, 896, 800},
    {384, 14, 14, 1, 1, 196, 384, 48, 64, 256, 224},
    {768,  7,  7, 1, 1,  49, 768, 96, 96, 128,  64},
  };
  static const long out_off[4] = {0, 4816896, 7225344, 8429568};

  // workspace layout (f16 elements, 256B aligned), sized for the max scale.
  // Activation buffers get +16 rows of slack so ceil-32 M-tiling can read
  // ghost rows safely (max padded use: scale3 = 800*768 elems << slot size).
  char* ws = (char*)d_ws;
  size_t off = 0;
  auto take = [&](size_t elems) { size_t o = off; off += ((elems * 2 + 255) / 256) * 256; return o; };
  size_t oXD = take(4816896 + 16 * 768), oXC = take(4816896 + 16 * 768);
  size_t oPJ = take(4816896), oAO = take(4816896 + 16 * 768);
  size_t oQ  = take(7340032), oK  = take(6553600), oVT = take(6553600);
  size_t oW[4]; for (int i = 0; i < 4; i++) oW[i] = take(589824);

  _Float16* XD = (_Float16*)(ws + oXD);
  _Float16* XC = (_Float16*)(ws + oXC);
  _Float16* PJ = (_Float16*)(ws + oPJ);
  _Float16* AO = (_Float16*)(ws + oAO);
  _Float16* Qb = (_Float16*)(ws + oQ);
  _Float16* Kb = (_Float16*)(ws + oK);
  _Float16* VTb = (_Float16*)(ws + oVT);

  for (int i = 0; i < 4; i++) {
    const Cfg& c = cfg[i];
    const float* xdet = (const float*)d_in[2 * i];
    const float* xctx = (const float*)d_in[2 * i + 1];
    int base = 8 + i * 8;
    const float* wq = (const float*)d_in[base + 0]; const float* bq = (const float*)d_in[base + 1];
    const float* wk = (const float*)d_in[base + 2]; const float* bk = (const float*)d_in[base + 3];
    const float* wv = (const float*)d_in[base + 4]; const float* bv = (const float*)d_in[base + 5];
    const float* wo = (const float*)d_in[base + 6]; const float* bo = (const float*)d_in[base + 7];
    _Float16* Wq = (_Float16*)(ws + oW[0]);
    _Float16* Wk = (_Float16*)(ws + oW[1]);
    _Float16* Wv = (_Float16*)(ws + oW[2]);
    _Float16* Wo = (_Float16*)(ws + oW[3]);

    int EE = c.E * c.E;
    int cb = (EE + 255) / 256;
    k_convert<<<cb, 256, 0, stream>>>(wq, Wq, EE);
    k_convert<<<cb, 256, 0, stream>>>(wk, Wk, EE);
    k_convert<<<cb, 256, 0, stream>>>(wv, Wv, EE);
    k_convert<<<cb, 256, 0, stream>>>(wo, Wo, EE);

    int M = B_ * c.N;
    long nx = (long)M * c.E;
    int fb = (int)((nx + 255) / 256);
    k_flatten<<<fb, 256, 0, stream>>>(xdet, XD, c.C, c.hh, c.ww, c.s1, c.s2, c.N, c.E);
    k_flatten<<<fb, 256, 0, stream>>>(xctx, XC, c.C, c.hh, c.ww, c.s1, c.s2, c.N, c.E);

    int waves = ((M + 31) / 32) * (c.E / 64);
    int gb = (waves + 7) / 8;

    long nq = (long)B_ * H_ * c.Npq * c.dp;
    long nk = (long)B_ * H_ * c.Npk * c.dp;
    int sbq = (int)((nq + 255) / 256);
    int sbk = (int)((nk + 255) / 256);

    // Q from context
    k_gemm_f16<<<gb, 256, 0, stream>>>(XC, Wq, bq, PJ, M, c.E, c.E);
    k_scatter_rows<<<sbq, 256, 0, stream>>>(PJ, Qb, c.Npq, c.dp, c.N, c.d, c.E);
    // K from detail
    k_gemm_f16<<<gb, 256, 0, stream>>>(XD, Wk, bk, PJ, M, c.E, c.E);
    k_scatter_rows<<<sbk, 256, 0, stream>>>(PJ, Kb, c.Npk, c.dp, c.N, c.d, c.E);
    // V from detail (stored transposed per head)
    k_gemm_f16<<<gb, 256, 0, stream>>>(XD, Wv, bv, PJ, M, c.E, c.E);
    k_scatter_vt<<<sbk, 256, 0, stream>>>(PJ, VTb, c.Npk, c.dp, c.N, c.d, c.E);

    // attention
    float scl = 1.0f / sqrtf((float)c.d);
    dim3 ag(c.Npq / 128, 1, B_ * H_);
    if (c.dp == 32)
      k_attn<32><<<ag, 256, 0, stream>>>(Qb, Kb, VTb, AO, c.N, c.Npq, c.Npk, c.d, c.E, scl);
    else if (c.dp == 64)
      k_attn<64><<<ag, 256, 0, stream>>>(Qb, Kb, VTb, AO, c.N, c.Npq, c.Npk, c.d, c.E, scl);
    else
      k_attn<96><<<ag, 256, 0, stream>>>(Qb, Kb, VTb, AO, c.N, c.Npq, c.Npk, c.d, c.E, scl);

    // output projection + unflatten + residual (f32)
    k_gemm_out<<<gb, 256, 0, stream>>>(AO, Wo, bo, xdet, (float*)d_out + out_off[i],
                                       M, c.E, c.E, c.C, c.hh, c.ww, c.s1, c.s2, c.N);
  }
}
